// AdaptiveFourierTransformGateLayer_90675349553327
// MI455X (gfx1250) — compile-verified, matrix operand order checked by reading, not run on hardware
//
#include <hip/hip_runtime.h>
#include <hip/hip_bf16.h>
#include <math.h>

// ---------------------------------------------------------------------------
// AdaptiveFourierTransformGateLayer for MI455X (gfx1250, wave32, WMMA).
//
// Shapes: B=2048, L=2048, NF=16, F=1024 (AC freq bins), FH=4096, E=experts.
// Heavy part: complex 2-layer MLP = 8 real GEMMs ~137 GFLOP -> f16 WMMA
// (v_wmma_f32_16x16x32_f16, f32 accumulate), with double-buffered
// GLOBAL_LOAD_ASYNC_TO_LDS_B128 staging (ASYNCcnt pipeline, 1 barrier/chunk).
//
// Workspace layout (80 MB total):
//   xr16, xi16      : f16 [2048,1024]    4 MB each
//   W1rt, W1it      : f16 [4096,1024]    8 MB each (w1 transposed)
//   W2rt, W2it      : f16 [1024,4096]    8 MB each (w2 transposed)
//   O1r,  O1i       : f16 [2048,4096]   16 MB each
//   amp             : f32 [2048,1024]    8 MB
// ---------------------------------------------------------------------------

typedef __attribute__((ext_vector_type(16))) _Float16 v16h;
typedef __attribute__((ext_vector_type(8)))  _Float16 v8h;
typedef __attribute__((ext_vector_type(8)))  float    v8f;

#define B_SZ   2048
#define L_SZ   2048
#define NF_SZ  16
#define F_SZ   1024
#define FH_SZ  4096

// -------------------------------------------------------------------------
// Kernel 0: fused FC (16->1) + 2048-pt radix-2 DIT FFT (ortho), per row.
// -------------------------------------------------------------------------
__global__ __launch_bounds__(256)
void fftfc_kernel(const float* __restrict__ x, const float* __restrict__ fc_w,
                  const float* __restrict__ fc_b,
                  _Float16* __restrict__ xr, _Float16* __restrict__ xi)
{
    __shared__ float sre[L_SZ];
    __shared__ float sim[L_SZ];
    const int tid = threadIdx.x;
    const int b   = blockIdx.x;
    const float* xrow = x + (size_t)b * (L_SZ * NF_SZ);

    float w[16];
    #pragma unroll
    for (int i = 0; i < 16; ++i) w[i] = fc_w[i];
    const float bias = fc_b[0];

    #pragma unroll
    for (int j = 0; j < 8; ++j) {
        int l = tid + 256 * j;
        const float4* p = reinterpret_cast<const float4*>(xrow + (size_t)l * NF_SZ);
        float4 a0 = p[0], a1 = p[1], a2 = p[2], a3 = p[3];
        float h = bias;
        h += a0.x*w[0]  + a0.y*w[1]  + a0.z*w[2]  + a0.w*w[3];
        h += a1.x*w[4]  + a1.y*w[5]  + a1.z*w[6]  + a1.w*w[7];
        h += a2.x*w[8]  + a2.y*w[9]  + a2.z*w[10] + a2.w*w[11];
        h += a3.x*w[12] + a3.y*w[13] + a3.z*w[14] + a3.w*w[15];
        int r = (int)(__brev((unsigned)l) >> 21);   // 11-bit reversal
        sre[r] = h;
        sim[r] = 0.0f;
    }

    for (int s = 0; s < 11; ++s) {
        __syncthreads();
        const int half = 1 << s;
        #pragma unroll
        for (int q = 0; q < 4; ++q) {
            int p  = tid + 256 * q;
            int j  = p & (half - 1);
            int i0 = ((p >> s) << (s + 1)) | j;
            int i1 = i0 + half;
            float ang = -3.14159265358979323846f * (float)j / (float)half;
            float sn, cs;
            __sincosf(ang, &sn, &cs);
            float re1 = sre[i1], im1 = sim[i1];
            float tr  = cs * re1 - sn * im1;
            float ti  = cs * im1 + sn * re1;
            float re0 = sre[i0], im0 = sim[i0];
            sre[i1] = re0 - tr;  sim[i1] = im0 - ti;
            sre[i0] = re0 + tr;  sim[i0] = im0 + ti;
        }
    }
    __syncthreads();

    const float sc = 0.022097086912f;   // 1/sqrt(2048)
    #pragma unroll
    for (int q = 0; q < 4; ++q) {
        int k = 1 + tid + 256 * q;
        size_t o = (size_t)b * F_SZ + (k - 1);
        xr[o] = (_Float16)(sre[k] * sc);
        xi[o] = (_Float16)(sim[k] * sc);
    }
}

// -------------------------------------------------------------------------
// Transpose + f32->f16 convert: in [R,C] f32 -> out [C,R] f16. 32x32 tiles.
// -------------------------------------------------------------------------
__global__ __launch_bounds__(256)
void transpose_cvt_kernel(const float* __restrict__ in, _Float16* __restrict__ out,
                          int R, int C)
{
    __shared__ float tile[32][33];
    const int tx = threadIdx.x, ty = threadIdx.y;
    const int bx = blockIdx.x * 32, by = blockIdx.y * 32;
    #pragma unroll
    for (int i = 0; i < 4; ++i)
        tile[ty + 8 * i][tx] = in[(size_t)(by + ty + 8 * i) * C + bx + tx];
    __syncthreads();
    #pragma unroll
    for (int i = 0; i < 4; ++i)
        out[(size_t)(bx + ty + 8 * i) * R + by + tx] = (_Float16)tile[tx][ty + 8 * i];
}

// -------------------------------------------------------------------------
// Async copy helpers (CDNA5: global -> LDS DMA, tracked by ASYNCcnt).
// GV mode: per-lane 32-bit LDS dest address VGPR + 64-bit global vaddr.
// -------------------------------------------------------------------------
__device__ __forceinline__ uint32_t lds_addr32(const void* p) {
    // generic pointer into the LDS aperture: low 32 bits are the LDS offset
    return (uint32_t)(uintptr_t)p;
}

__device__ __forceinline__ void async_copy_b128(const void* lds_dst,
                                                const _Float16* gsrc) {
    asm volatile("global_load_async_to_lds_b128 %0, %1, off"
                 :: "v"(lds_addr32(lds_dst)), "v"((uint64_t)(uintptr_t)gsrc)
                 : "memory");
}

__device__ __forceinline__ void wait_async0() {
    asm volatile("s_wait_asynccnt 0x0" ::: "memory");
}

// -------------------------------------------------------------------------
// Complex GEMM via WMMA:  O = A * W (+bias), A=[M,K] f16, W stored as Wt[N,K].
//   Or = Ar*Wr - Ai*Wi + br ;  Oi = Ai*Wr + Ar*Wi + bi
// MODE 0: relu -> f16 outputs.  MODE 1: amp = sqrt(Or^2+Oi^2) -> f32.
// WG tile 64x128, 8 waves as 2(M) x 4(N), 32x32 per wave (4 WMMA tiles).
// K in 32-chunks, double-buffered async global->LDS, 16 WMMA/wave/chunk.
// LDS: 2 bufs x (2x64x40 + 2x128x40) halfs = 60 KB.
// -------------------------------------------------------------------------
__device__ __forceinline__ v16h ld_frag(const _Float16* p) {
    // 16-bit A/B fragment: lane holds [row, kb..kb+7] and [row, kb+16..kb+23]
    union { v16h v; v8h h[2]; } u;
    u.h[0] = *(const v8h*)(p);
    u.h[1] = *(const v8h*)(p + 16);
    return u.v;
}

template <int MODE>
__global__ __launch_bounds__(256)
void cgemm_kernel(const _Float16* __restrict__ Ar, const _Float16* __restrict__ Ai,
                  const _Float16* __restrict__ Btr, const _Float16* __restrict__ Bti,
                  const float* __restrict__ br, const float* __restrict__ bi,
                  _Float16* __restrict__ oR, _Float16* __restrict__ oI,
                  float* __restrict__ ampOut, int M, int N, int K)
{
    __shared__ __align__(16) _Float16 sA[2][2][64 * 40];    // [buf][r/i]
    __shared__ __align__(16) _Float16 sB[2][2][128 * 40];

    const int tid  = threadIdx.x;
    const int wave = tid >> 5;
    const int lane = tid & 31;
    const int wm   = wave & 1;          // 0..1 : 32-row slab
    const int wn   = wave >> 1;         // 0..3 : 32-col slab
    const int lr   = lane & 15;         // fragment row within 16
    const int kb   = (lane >> 4) << 3;  // 0 or 8 : lane-half K base
    const int m0   = blockIdx.y * 64;
    const int n0   = blockIdx.x * 128;

    const v8f vzero = {0.f, 0.f, 0.f, 0.f, 0.f, 0.f, 0.f, 0.f};
    v8f accRR[2][2], accII[2][2], accRI[2][2];
    #pragma unroll
    for (int mi = 0; mi < 2; ++mi)
        #pragma unroll
        for (int ni = 0; ni < 2; ++ni) {
            accRR[mi][ni] = vzero; accII[mi][ni] = vzero; accRI[mi][ni] = vzero;
        }

    // Stage-loader index math (6 x b128 async copies per thread per chunk)
    const int arow = tid >> 2;                 // 0..63
    const int akg  = (tid & 3) << 3;           // 0,8,16,24
    const int brow = tid >> 1;                 // 0..127
    const int bkg  = (tid & 1) << 4;           // 0,16
    const size_t gaBase = (size_t)(m0 + arow) * K + akg;
    const size_t gbBase = (size_t)(n0 + brow) * K + bkg;
    const int aOff = arow * 40 + akg;
    const int bOff = brow * 40 + bkg;

    #define ISSUE_CHUNK(buf, k0)                                              \
        do {                                                                  \
            async_copy_b128(&sA[buf][0][aOff],     Ar  + gaBase + (k0));      \
            async_copy_b128(&sA[buf][1][aOff],     Ai  + gaBase + (k0));      \
            async_copy_b128(&sB[buf][0][bOff],     Btr + gbBase + (k0));      \
            async_copy_b128(&sB[buf][0][bOff + 8], Btr + gbBase + (k0) + 8);  \
            async_copy_b128(&sB[buf][1][bOff],     Bti + gbBase + (k0));      \
            async_copy_b128(&sB[buf][1][bOff + 8], Bti + gbBase + (k0) + 8);  \
        } while (0)

    const int nChunks = K >> 5;
    ISSUE_CHUNK(0, 0);                       // prologue: chunk 0 -> buf 0

    for (int kc = 0; kc < nChunks; ++kc) {
        const int buf = kc & 1;
        wait_async0();                       // own chunk-kc copies landed
        __syncthreads();                     // everyone's copies visible;
                                             // also: all waves done reading
                                             // buf (kc+1)&1 from chunk kc-1
        if (kc + 1 < nChunks) {
            const int k1 = (kc + 1) << 5;
            const int nb = buf ^ 1;
            ISSUE_CHUNK(nb, k1);             // overlap with compute below
        }

        v16h fAr[2], fAi[2], fBr[2], fBi[2];
        #pragma unroll
        for (int mi = 0; mi < 2; ++mi) {
            const int off = (wm * 32 + mi * 16 + lr) * 40 + kb;
            fAr[mi] = ld_frag(&sA[buf][0][off]);
            fAi[mi] = ld_frag(&sA[buf][1][off]);
        }
        #pragma unroll
        for (int ni = 0; ni < 2; ++ni) {
            const int off = (wn * 32 + ni * 16 + lr) * 40 + kb;
            fBr[ni] = ld_frag(&sB[buf][0][off]);
            fBi[ni] = ld_frag(&sB[buf][1][off]);
        }

        #pragma unroll
        for (int mi = 0; mi < 2; ++mi)
            #pragma unroll
            for (int ni = 0; ni < 2; ++ni) {
                accRR[mi][ni] = __builtin_amdgcn_wmma_f32_16x16x32_f16(
                    false, fAr[mi], false, fBr[ni], (short)0, accRR[mi][ni], false, false);
                accII[mi][ni] = __builtin_amdgcn_wmma_f32_16x16x32_f16(
                    false, fAi[mi], false, fBi[ni], (short)0, accII[mi][ni], false, false);
                accRI[mi][ni] = __builtin_amdgcn_wmma_f32_16x16x32_f16(
                    false, fAi[mi], false, fBr[ni], (short)0, accRI[mi][ni], false, false);
                accRI[mi][ni] = __builtin_amdgcn_wmma_f32_16x16x32_f16(
                    false, fAr[mi], false, fBi[ni], (short)0, accRI[mi][ni], false, false);
            }
    }
    #undef ISSUE_CHUNK

    // Epilogue. D layout: lane l -> n = l&15 ; vgpr r -> m = r + 8*(l>=16)
    const int mrow_off = (lane >> 4) << 3;
    #pragma unroll
    for (int mi = 0; mi < 2; ++mi)
        #pragma unroll
        for (int ni = 0; ni < 2; ++ni) {
            const int ng  = n0 + wn * 32 + ni * 16 + lr;
            const float vbr = br[ng];
            const float vbi = bi[ng];
            #pragma unroll
            for (int r = 0; r < 8; ++r) {
                const int mg = m0 + wm * 32 + mi * 16 + r + mrow_off;
                float orv = accRR[mi][ni][r] - accII[mi][ni][r] + vbr;
                float oiv = accRI[mi][ni][r] + vbi;
                if (MODE == 0) {
                    oR[(size_t)mg * N + ng] = (_Float16)fmaxf(orv, 0.0f);
                    oI[(size_t)mg * N + ng] = (_Float16)fmaxf(oiv, 0.0f);
                } else {
                    ampOut[(size_t)mg * N + ng] = sqrtf(orv * orv + oiv * oiv);
                }
            }
        }
}

// -------------------------------------------------------------------------
// Noisy top-3 gating. One block per batch row.
// -------------------------------------------------------------------------
__global__ __launch_bounds__(256)
void gate_kernel(const float* __restrict__ amp, const float* __restrict__ w_gate,
                 const float* __restrict__ w_noise, const float* __restrict__ eps,
                 const int* __restrict__ training, float* __restrict__ gates, int E)
{
    __shared__ float ampL[F_SZ];
    __shared__ float logitsL[256];
    const int tid = threadIdx.x;
    const int b   = blockIdx.x;

    #pragma unroll
    for (int q = 0; q < 4; ++q)
        ampL[tid + 256 * q] = amp[(size_t)b * F_SZ + tid + 256 * q];
    __syncthreads();

    const int train = training[0];
    for (int e = tid; e < E; e += 256) {
        float c = 0.f, nz = 0.f;
        for (int k = 0; k < F_SZ; ++k) {
            const float a = ampL[k];
            c  += a * w_gate [(size_t)k * E + e];
            nz += a * w_noise[(size_t)k * E + e];
        }
        float logit = c;
        if (train != 0) {
            float sp = (nz > 20.f) ? nz : log1pf(expf(nz));   // softplus
            logit += eps[(size_t)b * E + e] * (sp + 0.01f);
        }
        logitsL[e] = logit;
        gates[(size_t)b * E + e] = 0.0f;   // zero-fill row
    }
    __syncthreads();

    if (tid == 0) {
        int   idx[3] = {-1, -1, -1};
        float val[3];
        #pragma unroll
        for (int p = 0; p < 3; ++p) {
            float best = -INFINITY; int bi_ = -1;
            for (int e = 0; e < E; ++e) {
                if (e == idx[0] || e == idx[1]) continue;
                float v = logitsL[e];
                if (v > best) { best = v; bi_ = e; }
            }
            idx[p] = bi_; val[p] = best;
        }
        const float m  = val[0];
        const float e0 = 1.0f;
        const float e1 = expf(val[1] - m);
        const float e2 = expf(val[2] - m);
        const float s  = e0 + e1 + e2;
        gates[(size_t)b * E + idx[0]] = e0 / s;
        gates[(size_t)b * E + idx[1]] = e1 / s;
        gates[(size_t)b * E + idx[2]] = e2 / s;
    }
}

// -------------------------------------------------------------------------
extern "C" void kernel_launch(void* const* d_in, const int* in_sizes, int n_in,
                              void* d_out, int out_size, void* d_ws, size_t ws_size,
                              hipStream_t stream) {
    const float* x       = (const float*)d_in[0];
    const float* fc_w    = (const float*)d_in[1];
    const float* fc_b    = (const float*)d_in[2];
    const float* w1      = (const float*)d_in[3];   // [2, 1024, 4096]
    const float* b1      = (const float*)d_in[4];   // [2, 4096]
    const float* w2      = (const float*)d_in[5];   // [2, 4096, 1024]
    const float* b2      = (const float*)d_in[6];   // [2, 1024]
    const float* w_gate  = (const float*)d_in[7];   // [1024, E]
    const float* w_noise = (const float*)d_in[8];
    const float* eps     = (const float*)d_in[9];   // [B, E]
    const int*   training= (const int*)  d_in[10];
    const int E = in_sizes[7] / F_SZ;
    float* gates = (float*)d_out;

    char* ws = (char*)d_ws;
    size_t off = 0;
    _Float16* xr16 = (_Float16*)(ws + off); off += (size_t)B_SZ * F_SZ  * 2;
    _Float16* xi16 = (_Float16*)(ws + off); off += (size_t)B_SZ * F_SZ  * 2;
    _Float16* W1rt = (_Float16*)(ws + off); off += (size_t)FH_SZ * F_SZ * 2;
    _Float16* W1it = (_Float16*)(ws + off); off += (size_t)FH_SZ * F_SZ * 2;
    _Float16* W2rt = (_Float16*)(ws + off); off += (size_t)F_SZ * FH_SZ * 2;
    _Float16* W2it = (_Float16*)(ws + off); off += (size_t)F_SZ * FH_SZ * 2;
    _Float16* O1r  = (_Float16*)(ws + off); off += (size_t)B_SZ * FH_SZ * 2;
    _Float16* O1i  = (_Float16*)(ws + off); off += (size_t)B_SZ * FH_SZ * 2;
    float*    ampb = (float*)   (ws + off); off += (size_t)B_SZ * F_SZ  * 4;

    // 1) FC + FFT
    fftfc_kernel<<<B_SZ, 256, 0, stream>>>(x, fc_w, fc_b, xr16, xi16);

    // 2) Weight transpose + f16 convert
    transpose_cvt_kernel<<<dim3(FH_SZ / 32, F_SZ / 32), dim3(32, 8), 0, stream>>>(
        w1,                        W1rt, F_SZ, FH_SZ);
    transpose_cvt_kernel<<<dim3(FH_SZ / 32, F_SZ / 32), dim3(32, 8), 0, stream>>>(
        w1 + (size_t)F_SZ * FH_SZ, W1it, F_SZ, FH_SZ);
    transpose_cvt_kernel<<<dim3(F_SZ / 32, FH_SZ / 32), dim3(32, 8), 0, stream>>>(
        w2,                        W2rt, FH_SZ, F_SZ);
    transpose_cvt_kernel<<<dim3(F_SZ / 32, FH_SZ / 32), dim3(32, 8), 0, stream>>>(
        w2 + (size_t)FH_SZ * F_SZ, W2it, FH_SZ, F_SZ);

    // 3) Layer 1: [2048,1024] x [1024,4096] complex, relu -> f16
    cgemm_kernel<0><<<dim3(FH_SZ / 128, B_SZ / 64), 256, 0, stream>>>(
        xr16, xi16, W1rt, W1it, b1, b1 + FH_SZ, O1r, O1i, nullptr,
        B_SZ, FH_SZ, F_SZ);

    // 4) Layer 2: [2048,4096] x [4096,1024] complex, fused amp -> f32
    cgemm_kernel<1><<<dim3(F_SZ / 128, B_SZ / 64), 256, 0, stream>>>(
        O1r, O1i, W2rt, W2it, b2, b2 + F_SZ, nullptr, nullptr, ampb,
        B_SZ, F_SZ, FH_SZ);

    // 5) Noisy top-3 gating
    gate_kernel<<<B_SZ, 256, 0, stream>>>(ampb, w_gate, w_noise, eps, training,
                                          gates, E);
}
